// Bond2AtomLayer_41532333752517
// MI455X (gfx1250) — matrix operand order repaired and checked by reading, not compiled, not run on hardware
//
#include <hip/hip_runtime.h>
#include <hip/hip_bf16.h>
#include <stdint.h>

// ---------------------------------------------------------------------------
// Bond2AtomLayer for MI455X (gfx1250, wave32, WMMA).
// Memory-bound (~1.1 GB @ 23.3 TB/s): GEMMs on v_wmma_f32_16x16x32_bf16,
// V-projection fused into the edge->node scatter, weights pre-converted to
// bf16 once, FFN activations stored bf16, FFN A-tiles staged to LDS via the
// CDNA5 async global->LDS path when available.
// ---------------------------------------------------------------------------

#define N_NODES 50000
#define N_EDGES 800000
#define D_MODEL 128
#define N_HEAD  8
#define D_HEAD  16
#define D_FF    256
#define LN_EPS  1e-5f

typedef __bf16 bf16_t;
typedef __attribute__((ext_vector_type(16))) __bf16 v16bf;
typedef __attribute__((ext_vector_type(4)))  __bf16 v4bf;
typedef __attribute__((ext_vector_type(8)))  float  v8f;
typedef __attribute__((vector_size(16)))     int    b128_t;

#if defined(__has_builtin)
#if __has_builtin(__builtin_amdgcn_global_load_async_to_lds_b128)
#define ASYNC_STAGE 1
#endif
#endif

// 16-byte global -> LDS copy; async (ASYNCcnt-tracked) when available.
__device__ __forceinline__ void async_copy16(void* ldst, const void* gsrc) {
#ifdef ASYNC_STAGE
  __builtin_amdgcn_global_load_async_to_lds_b128(
      (__attribute__((address_space(1))) b128_t*)(uintptr_t)gsrc,
      (__attribute__((address_space(3))) b128_t*)(uint32_t)(uintptr_t)ldst,
      0, 0);
#else
  *(uint4*)ldst = *(const uint4*)gsrc;
#endif
}

__device__ __forceinline__ void wait_async() {
#ifdef ASYNC_STAGE
#if __has_builtin(__builtin_amdgcn_s_wait_asynccnt)
  __builtin_amdgcn_s_wait_asynccnt(0);
#else
  asm volatile("s_wait_asynccnt 0x0" ::: "memory");
#endif
#endif
}

// ---------------- WMMA fragment helpers (contiguous k order: lane half h holds k = 16h+s) ------------

// A fragment: 16x32 bf16 tile from LDS (row-major, leading dim ldk).
__device__ __forceinline__ v16bf a_frag_lds(const bf16_t* __restrict__ As, int ldk,
                                            int kb, int lane) {
  int m    = lane & 15;
  int koff = (lane >> 4) ? 16 : 0;
  const bf16_t* p = As + m * ldk + kb + koff;
  v16bf a;
#pragma unroll
  for (int h = 0; h < 8; ++h) a[h] = p[h];
#pragma unroll
  for (int h = 0; h < 8; ++h) a[8 + h] = p[8 + h];
  return a;
}

// B fragment for out = X @ W^T from a pre-converted bf16 weight matrix.
// lanes 0-15 hold K = kb+0..15 of column n=lane; lanes 16-31 hold K = kb+16..31.
// (c0+n)*ldw + kbase is a multiple of 16 elements -> 32-byte aligned vector load.
__device__ __forceinline__ v16bf bT_frag_bf(const bf16_t* __restrict__ W, int ldw,
                                            int c0, int kb, int lane) {
  int n     = lane & 15;
  int kbase = kb + ((lane >> 4) ? 16 : 0);
  return *(const v16bf*)(W + (size_t)(c0 + n) * ldw + kbase);
}

__device__ __forceinline__ v8f wmma_bf16(v16bf a, v16bf b, v8f c) {
  return __builtin_amdgcn_wmma_f32_16x16x32_bf16(false, a, false, b, (short)0, c,
                                                 false, false);
}

__device__ __forceinline__ float wave_sum32(float v) {
#pragma unroll
  for (int m = 16; m >= 1; m >>= 1) v += __shfl_xor(v, m, 32);
  return v;
}

__device__ __forceinline__ void atomicMaxFloat(float* addr, float val) {
  if (!(val < 0.0f)) atomicMax((int*)addr, __float_as_int(val));
  else               atomicMin((unsigned int*)addr, __float_as_uint(val));
}

// ---------------------------- kernels ---------------------------------------

__global__ void __launch_bounds__(256) init_kernel(float* __restrict__ m,
                                                   float* __restrict__ s,
                                                   float* __restrict__ ft) {
  int i = blockIdx.x * 256 + threadIdx.x;
  if (i < N_NODES * N_HEAD) { m[i] = -INFINITY; s[i] = 0.0f; }
  if (i < N_NODES * D_MODEL) ft[i] = 0.0f;
}

// One-time f32 -> bf16 conversion of all GEMM weights into workspace.
// Layout: [0,16384) Wk | [,32768) Wq | [,49152) Wv | [,81920) W1 | [,114688) W2
__global__ void __launch_bounds__(256) wconv_kernel(const float* __restrict__ Wk,
    const float* __restrict__ Wq, const float* __restrict__ Wv,
    const float* __restrict__ W1, const float* __restrict__ W2,
    bf16_t* __restrict__ wbf) {
  int i = blockIdx.x * 256 + threadIdx.x;
  float v;
  if      (i < 16384) v = Wk[i];
  else if (i < 32768) v = Wq[i - 16384];
  else if (i < 49152) v = Wv[i - 32768];
  else if (i < 81920) v = W1[i - 49152];
  else                v = W2[i - 81920];
  wbf[i] = (bf16_t)v;
}

// k = X @ Wk^T, q = X @ Wq^T  (3125 blocks of 16 rows; wave w -> col tile w)
__global__ void __launch_bounds__(256) proj_kq_kernel(const float* __restrict__ x,
    const bf16_t* __restrict__ Wkb, const bf16_t* __restrict__ Wqb,
    float* __restrict__ kout, float* __restrict__ qout) {
  __shared__ bf16_t As[16 * 128];
  int rowbase = blockIdx.x * 16;
  int t = threadIdx.x;
#pragma unroll
  for (int i = 0; i < 8; ++i) {
    int j = t + i * 256;
    As[j] = (bf16_t)x[(size_t)rowbase * 128 + j];
  }
  __syncthreads();
  int wave = t >> 5, lane = t & 31;
  int c0 = wave * 16;
  v8f ck = {}, cq = {};
#pragma unroll
  for (int kb = 0; kb < 128; kb += 32) {
    v16bf a  = a_frag_lds(As, 128, kb, lane);
    v16bf bk = bT_frag_bf(Wkb, 128, c0, kb, lane);
    v16bf bq = bT_frag_bf(Wqb, 128, c0, kb, lane);
    ck = wmma_bf16(a, bk, ck);
    cq = wmma_bf16(a, bq, cq);
  }
  int n = lane & 15, moff = (lane >> 4) ? 8 : 0;
#pragma unroll
  for (int r = 0; r < 8; ++r) {
    int row = rowbase + r + moff;
    kout[(size_t)row * 128 + c0 + n] = ck[r];
    qout[(size_t)row * 128 + c0 + n] = cq[r];
  }
}

// logits[e,h] = dot(k[src[e],h], q[dst[e],h]) / 4 + basic_attn[e]*W_dis[h]
__global__ void __launch_bounds__(256) edge_logits_kernel(const float* __restrict__ k,
    const float* __restrict__ q, const float* __restrict__ ba,
    const float* __restrict__ Wdis, const int* __restrict__ src,
    const int* __restrict__ dst, float* __restrict__ logits,
    float* __restrict__ mbuf) {
  int idx = blockIdx.x * 256 + threadIdx.x;
  if (idx >= N_EDGES * N_HEAD) return;
  int e = idx >> 3, h = idx & 7;
  int si = src[e], di = dst[e];
  const float4* kp = (const float4*)(k + (size_t)si * 128 + h * 16);
  const float4* qp = (const float4*)(q + (size_t)di * 128 + h * 16);
  float acc = 0.0f;
#pragma unroll
  for (int i = 0; i < 4; ++i) {
    float4 a = kp[i], b = qp[i];
    acc += a.x * b.x + a.y * b.y + a.z * b.z + a.w * b.w;
  }
  float lg = acc * 0.25f + ba[e] * Wdis[h];
  logits[idx] = lg;
  atomicMaxFloat(mbuf + (size_t)di * 8 + h, lg);
}

// p = exp(logit - m[dst]); s[dst] += p
__global__ void __launch_bounds__(256) edge_exp_kernel(float* __restrict__ logits,
    const float* __restrict__ mbuf, float* __restrict__ sbuf,
    const int* __restrict__ dst) {
  int idx = blockIdx.x * 256 + threadIdx.x;
  if (idx >= N_EDGES * N_HEAD) return;
  int e = idx >> 3, h = idx & 7;
  int di = dst[e];
  float p = __expf(logits[idx] - mbuf[(size_t)di * 8 + h]);
  logits[idx] = p;
  atomicAdd(sbuf + (size_t)di * 8 + h, p);
}

// a = p / s[dst]  (normalize once; scatter becomes pure multiply)
__global__ void __launch_bounds__(256) edge_norm_kernel(float* __restrict__ pbuf,
    const float* __restrict__ sbuf, const int* __restrict__ dst) {
  int idx = blockIdx.x * 256 + threadIdx.x;
  if (idx >= N_EDGES * N_HEAD) return;
  int e = idx >> 3, h = idx & 7;
  pbuf[idx] = pbuf[idx] / sbuf[(size_t)dst[e] * 8 + h];
}

// ft[dst] += a * (bond @ Wv^T): 50000 blocks of 16 edges; wave h -> head h.
__global__ void __launch_bounds__(256) scatter_v_kernel(const float* __restrict__ bond,
    const bf16_t* __restrict__ Wvb, const float* __restrict__ abuf,
    const int* __restrict__ dst, float* __restrict__ ft) {
  __shared__ bf16_t As[16 * 128];
  int ebase = blockIdx.x * 16;
  int t = threadIdx.x;
#pragma unroll
  for (int i = 0; i < 8; ++i) {
    int j = t + i * 256;
    As[j] = (bf16_t)bond[(size_t)ebase * 128 + j];
  }
  __syncthreads();
  int wave = t >> 5, lane = t & 31;
  int h = wave;
  v8f c = {};
#pragma unroll
  for (int kb = 0; kb < 128; kb += 32) {
    v16bf a = a_frag_lds(As, 128, kb, lane);
    v16bf b = bT_frag_bf(Wvb, 128, h * 16, kb, lane);
    c = wmma_bf16(a, b, c);
  }
  int n = lane & 15, moff = (lane >> 4) ? 8 : 0;
#pragma unroll
  for (int r = 0; r < 8; ++r) {
    int e = ebase + r + moff;
    int d = dst[e];
    float aw = abuf[(size_t)e * 8 + h];
    atomicAdd(ft + (size_t)d * 128 + h * 16 + n, c[r] * aw);
  }
}

// beta gating + affine LN1; one wave per node. he (=ft) updated in place with
// the gated residual (f32); hnorm written as bf16 (WMMA A-operand downstream).
__global__ void __launch_bounds__(256) gate_ln_kernel(float* __restrict__ he,
    const float* __restrict__ x, const float* __restrict__ Wb,
    const float* __restrict__ g, const float* __restrict__ b,
    bf16_t* __restrict__ hnorm) {
  int node = blockIdx.x * 8 + (threadIdx.x >> 5);
  int lane = threadIdx.x & 31;
  float4 hv = ((const float4*)(he + (size_t)node * 128))[lane];
  float4 xv = ((const float4*)(x + (size_t)node * 128))[lane];
  const float4* wb4 = (const float4*)Wb;
  float4 w0 = wb4[lane], w1 = wb4[32 + lane], w2 = wb4[64 + lane];
  float part = hv.x * w0.x + hv.y * w0.y + hv.z * w0.z + hv.w * w0.w
             + xv.x * w1.x + xv.y * w1.y + xv.z * w1.z + xv.w * w1.w
             + (hv.x - xv.x) * w2.x + (hv.y - xv.y) * w2.y
             + (hv.z - xv.z) * w2.z + (hv.w - xv.w) * w2.w;
  float tot  = wave_sum32(part);
  float beta = 1.0f / (1.0f + __expf(-tot));
  float4 gt;
  gt.x = beta * xv.x + (1.0f - beta) * hv.x;
  gt.y = beta * xv.y + (1.0f - beta) * hv.y;
  gt.z = beta * xv.z + (1.0f - beta) * hv.z;
  gt.w = beta * xv.w + (1.0f - beta) * hv.w;
  float s1 = gt.x + gt.y + gt.z + gt.w;
  float s2 = gt.x * gt.x + gt.y * gt.y + gt.z * gt.z + gt.w * gt.w;
  s1 = wave_sum32(s1);
  s2 = wave_sum32(s2);
  float mu   = s1 * (1.0f / 128.0f);
  float var  = s2 * (1.0f / 128.0f) - mu * mu;
  float rstd = rsqrtf(var + LN_EPS);
  float4 gm = ((const float4*)g)[lane], bb = ((const float4*)b)[lane];
  v4bf y;
  y[0] = (bf16_t)((gt.x - mu) * rstd * gm.x + bb.x);
  y[1] = (bf16_t)((gt.y - mu) * rstd * gm.y + bb.y);
  y[2] = (bf16_t)((gt.z - mu) * rstd * gm.z + bb.z);
  y[3] = (bf16_t)((gt.w - mu) * rstd * gm.w + bb.w);
  ((float4*)(he + (size_t)node * 128))[lane] = gt;
  ((v4bf*)(hnorm + (size_t)node * 128))[lane] = y;
}

// hidden = relu(hnorm @ W1^T)  [N,128] x [256,128]^T -> [N,256] (bf16 out)
__global__ void __launch_bounds__(256) ffn1_kernel(const bf16_t* __restrict__ hnorm,
    const bf16_t* __restrict__ W1b, bf16_t* __restrict__ hidden) {
  __shared__ bf16_t As[16 * 128];
  int rowbase = blockIdx.x * 16;
  int t = threadIdx.x;
  // 16x128 bf16 tile = 4 KB: each thread async-copies 16 bytes.
  async_copy16(As + t * 8, hnorm + (size_t)rowbase * 128 + t * 8);
  wait_async();
  __syncthreads();
  int wave = t >> 5, lane = t & 31;
  v8f c0 = {}, c1 = {};
#pragma unroll
  for (int kb = 0; kb < 128; kb += 32) {
    v16bf a  = a_frag_lds(As, 128, kb, lane);
    v16bf b0 = bT_frag_bf(W1b, 128, wave * 16, kb, lane);
    v16bf b1 = bT_frag_bf(W1b, 128, (wave + 8) * 16, kb, lane);
    c0 = wmma_bf16(a, b0, c0);
    c1 = wmma_bf16(a, b1, c1);
  }
  int n = lane & 15, moff = (lane >> 4) ? 8 : 0;
#pragma unroll
  for (int r = 0; r < 8; ++r) {
    int row = rowbase + r + moff;
    hidden[(size_t)row * 256 + wave * 16 + n]       = (bf16_t)fmaxf(c0[r], 0.0f);
    hidden[(size_t)row * 256 + (wave + 8) * 16 + n] = (bf16_t)fmaxf(c1[r], 0.0f);
  }
}

// out = LN(hidden @ W2^T + he_residual)  (non-affine LN)
__global__ void __launch_bounds__(256) ffn2_ln_kernel(const bf16_t* __restrict__ hidden,
    const bf16_t* __restrict__ W2b, const float* __restrict__ hres,
    float* __restrict__ out) {
  __shared__ bf16_t As[16 * 256];
  __shared__ float  tile[16 * 128];
  int rowbase = blockIdx.x * 16;
  int t = threadIdx.x;
  // 16x256 bf16 tile = 8 KB: each thread async-copies 2 x 16 bytes.
  {
    const bf16_t* gsrc = hidden + (size_t)rowbase * 256;
    async_copy16(As + t * 8,        gsrc + t * 8);
    async_copy16(As + 2048 + t * 8, gsrc + 2048 + t * 8);
  }
  wait_async();
  __syncthreads();
  int wave = t >> 5, lane = t & 31;
  int c0 = wave * 16;
  v8f c = {};
#pragma unroll
  for (int kb = 0; kb < 256; kb += 32) {
    v16bf a = a_frag_lds(As, 256, kb, lane);
    v16bf b = bT_frag_bf(W2b, 256, c0, kb, lane);
    c = wmma_bf16(a, b, c);
  }
  int n = lane & 15, moff = (lane >> 4) ? 8 : 0;
#pragma unroll
  for (int r = 0; r < 8; ++r) {
    int ml = r + moff;
    tile[ml * 128 + c0 + n] =
        c[r] + hres[(size_t)(rowbase + ml) * 128 + c0 + n];
  }
  __syncthreads();
  // Row-wise non-affine LN: 16 threads per row, 8 elements each.
  int row = t >> 4, seg = t & 15;
  const float* rp = &tile[row * 128 + seg * 8];
  float s1 = 0.0f, s2 = 0.0f;
#pragma unroll
  for (int i = 0; i < 8; ++i) { float v = rp[i]; s1 += v; s2 += v * v; }
#pragma unroll
  for (int m = 8; m >= 1; m >>= 1) {
    s1 += __shfl_xor(s1, m, 16);
    s2 += __shfl_xor(s2, m, 16);
  }
  float mu   = s1 * (1.0f / 128.0f);
  float var  = s2 * (1.0f / 128.0f) - mu * mu;
  float rstd = rsqrtf(var + LN_EPS);
  float* op = out + (size_t)(rowbase + row) * 128 + seg * 8;
#pragma unroll
  for (int i = 0; i < 8; ++i) op[i] = (rp[i] - mu) * rstd;
}

// ------------------------------ launcher -------------------------------------

extern "C" void kernel_launch(void* const* d_in, const int* in_sizes, int n_in,
                              void* d_out, int out_size, void* d_ws, size_t ws_size,
                              hipStream_t stream) {
  const float* bond  = (const float*)d_in[0];
  const float* node  = (const float*)d_in[1];
  const float* ba    = (const float*)d_in[2];
  const float* Wk    = (const float*)d_in[3];
  const float* Wq    = (const float*)d_in[4];
  const float* Wv    = (const float*)d_in[5];
  const float* Wdis  = (const float*)d_in[6];
  const float* Wbeta = (const float*)d_in[7];
  const float* W1    = (const float*)d_in[8];
  const float* W2    = (const float*)d_in[9];
  const float* ln1g  = (const float*)d_in[10];
  const float* ln1b  = (const float*)d_in[11];
  const int*   src   = (const int*)d_in[12];
  const int*   dst   = (const int*)d_in[13];
  float* out = (float*)d_out;

  // f32 region
  float* ws    = (float*)d_ws;
  float* kbuf  = ws;                                   // N*128
  float* qbuf  = kbuf  + (size_t)N_NODES * D_MODEL;    // N*128
  float* pbuf  = qbuf  + (size_t)N_NODES * D_MODEL;    // E*8 (logits -> p -> a)
  float* mbuf  = pbuf  + (size_t)N_EDGES * N_HEAD;     // N*8
  float* sbuf  = mbuf  + (size_t)N_NODES * N_HEAD;     // N*8
  float* ftbuf = sbuf  + (size_t)N_NODES * N_HEAD;     // N*128 (ft -> gated he)
  // bf16 region (32-byte aligned: all preceding counts are multiples of 8)
  bf16_t* hnorm = (bf16_t*)(ftbuf + (size_t)N_NODES * D_MODEL); // N*128 bf16
  bf16_t* hid   = hnorm + (size_t)N_NODES * D_MODEL;            // N*256 bf16
  bf16_t* wbf   = hid   + (size_t)N_NODES * D_FF;               // 114688 bf16
  bf16_t* Wkb = wbf;
  bf16_t* Wqb = wbf + 16384;
  bf16_t* Wvb = wbf + 32768;
  bf16_t* W1b = wbf + 49152;
  bf16_t* W2b = wbf + 81920;

  const int edgework = (N_EDGES * N_HEAD + 255) / 256;  // 25000
  const int nodetile = N_NODES / 16;                    // 3125
  const int edgetile = N_EDGES / 16;                    // 50000

  init_kernel<<<(N_NODES * D_MODEL + 255) / 256, 256, 0, stream>>>(mbuf, sbuf, ftbuf);
  wconv_kernel<<<114688 / 256, 256, 0, stream>>>(Wk, Wq, Wv, W1, W2, wbf);
  proj_kq_kernel<<<nodetile, 256, 0, stream>>>(node, Wkb, Wqb, kbuf, qbuf);
  edge_logits_kernel<<<edgework, 256, 0, stream>>>(kbuf, qbuf, ba, Wdis, src, dst,
                                                   pbuf, mbuf);
  edge_exp_kernel<<<edgework, 256, 0, stream>>>(pbuf, mbuf, sbuf, dst);
  edge_norm_kernel<<<edgework, 256, 0, stream>>>(pbuf, sbuf, dst);
  scatter_v_kernel<<<edgetile, 256, 0, stream>>>(bond, Wvb, pbuf, dst, ftbuf);
  gate_ln_kernel<<<N_NODES / 8, 256, 0, stream>>>(ftbuf, node, Wbeta, ln1g, ln1b,
                                                  hnorm);
  ffn1_kernel<<<nodetile, 256, 0, stream>>>(hnorm, W1b, hid);
  ffn2_ln_kernel<<<nodetile, 256, 0, stream>>>(hid, W2b, ftbuf, out);
}
